// Attention_91061896610676
// MI455X (gfx1250) — matrix operand order repaired and checked by reading, not compiled, not run on hardware
//
#include <hip/hip_runtime.h>

typedef unsigned short u16;
typedef __attribute__((ext_vector_type(16))) __bf16 v16bf;
typedef __attribute__((ext_vector_type(8)))  float  v8f;

#define DEVI __device__ __forceinline__

static constexpr int BN   = 16384;
static constexpr int Hd   = 1024;
static constexpr int OUTN = 128;

// ---------- helpers ----------
DEVI u16 f2bf(float f) {
  union { float f; unsigned int u; } x; x.f = f;
  unsigned int u = x.u;
  u += 0x7FFFu + ((u >> 16) & 1u);          // round-to-nearest-even
  return (u16)(u >> 16);
}
DEVI float b2f(u16 h) {
  union { unsigned int u; float f; } x; x.u = ((unsigned int)h) << 16;
  return x.f;
}
DEVI void st_bf4(u16* dst, float4 v) {      // 4 f32 -> 4 bf16, one 8B store
  uint2 p;
  p.x = (unsigned int)f2bf(v.x) | ((unsigned int)f2bf(v.y) << 16);
  p.y = (unsigned int)f2bf(v.z) | ((unsigned int)f2bf(v.w) << 16);
  *(uint2*)dst = p;
}
DEVI v8f vzero8() { v8f z = {0.f,0.f,0.f,0.f,0.f,0.f,0.f,0.f}; return z; }

union FragAB { v16bf v; uint4 q[2]; };
DEVI uint4 lds_ld16(const u16* p) { return *(const uint4*)p; }

// ---------- CDNA5 async global->LDS (ASYNCcnt-tracked, no VGPR round trip) ----------
DEVI void async_ld16(const u16* lds_dst, const u16* gsrc) {
  unsigned lds = (unsigned)(size_t)(void*)lds_dst;                 // low 32 bits = LDS byte addr
  unsigned long long ga = (unsigned long long)(size_t)(const void*)gsrc;
  asm volatile("global_load_async_to_lds_b128 %0, %1, off"
               :: "v"(lds), "v"(ga) : "memory");
}
DEVI void wait_async0() {
#if __has_builtin(__builtin_amdgcn_s_wait_asynccnt)
  __builtin_amdgcn_s_wait_asynccnt(0);
#else
  asm volatile("s_wait_asynccnt 0" ::: "memory");
#endif
}
DEVI void wait_async4() {
#if __has_builtin(__builtin_amdgcn_s_wait_asynccnt)
  __builtin_amdgcn_s_wait_asynccnt(4);
#else
  asm volatile("s_wait_asynccnt 4" ::: "memory");
#endif
}

// Stage a [128 rows x 64 elem] bf16 slab (dst stride 72) from row-major src
// (row stride Hd) using async loads: 4 x b128 per thread.
DEVI void stage_slab_async(u16* dst, const u16* src, int tid) {
#pragma unroll
  for (int g4 = tid; g4 < 128 * 8; g4 += 256) {
    int r = g4 >> 3, c8 = g4 & 7;
    async_ld16(&dst[r * 72 + c8 * 8], src + (size_t)r * Hd + c8 * 8);
  }
}

// One K=32 WMMA step for a 16(M) x 64(N) wave strip.
// Gather all fragments first, then issue the 4 WMMAs back-to-back.
template<int AS, int BS>
DEVI void wmma_k32(const u16* Asrc, const u16* Bsrc, int ks, int lane, v8f acc[4]) {
  const int g = lane >> 4, mn = lane & 15;
  FragAB a;
  a.q[0] = lds_ld16(Asrc + mn * AS + ks + 8 * g);        // K = ks+8g .. +8
  a.q[1] = lds_ld16(Asrc + mn * AS + ks + 16 + 8 * g);   // K = ks+16+8g .. +8
  FragAB b[4];
#pragma unroll
  for (int s = 0; s < 4; ++s) {
    const u16* bp = Bsrc + (s * 16 + mn) * BS + ks + 8 * g;  // K = ks+8g .. +8, ks+16+8g .. +8
    b[s].q[0] = lds_ld16(bp);
    b[s].q[1] = lds_ld16(bp + 16);
  }
#pragma unroll
  for (int s = 0; s < 4; ++s)
    acc[s] = __builtin_amdgcn_wmma_f32_16x16x32_bf16(
        false, a.v, false, b[s].v, (short)0, acc[s], false, false);
}

// ---------- prep: transpose Wv, Wfc into bf16 [n][k] ----------
__global__ __launch_bounds__(256) void k_prep_transpose(
    const float* __restrict__ Wv, const float* __restrict__ Wfc,
    u16* __restrict__ Wvt, u16* __restrict__ Wfct) {
  int idx = blockIdx.x * 256 + threadIdx.x;
  if (idx < Hd * Hd) {
    int k = idx >> 10, n = idx & 1023;
    Wvt[n * Hd + k] = f2bf(Wv[idx]);                 // Wvt[n][k] = Wv[k][n]
  } else {
    int j = idx - Hd * Hd;
    if (j < Hd * OUTN) {
      int k = j >> 7, o = j & 127;
      Wfct[o * Hd + k] = f2bf(Wfc[j]);               // Wfct[o][k] = Wfc[k][o]
    }
  }
}

// ---------- prep: g = Wk@bq, h = Wq@bk, c = bq.bk ----------
__global__ __launch_bounds__(256) void k_prep_bias(
    const float* __restrict__ Wq, const float* __restrict__ Wk,
    const float* __restrict__ bq, const float* __restrict__ bk,
    float* __restrict__ g, float* __restrict__ h, float* __restrict__ c) {
  int idx = blockIdx.x * 256 + threadIdx.x;   // grid covers 2*Hd
  if (idx < Hd) {
    const float4* row = (const float4*)(Wk + (size_t)idx * Hd);
    const float4* b4  = (const float4*)bq;
    float s = 0.f;
    for (int j = 0; j < Hd / 4; ++j) {
      float4 a = row[j], bb = b4[j];
      s += a.x * bb.x + a.y * bb.y + a.z * bb.z + a.w * bb.w;
    }
    g[idx] = s;
  } else if (idx < 2 * Hd) {
    int k = idx - Hd;
    const float4* row = (const float4*)(Wq + (size_t)k * Hd);
    const float4* b4  = (const float4*)bk;
    float s = 0.f;
    for (int j = 0; j < Hd / 4; ++j) {
      float4 a = row[j], bb = b4[j];
      s += a.x * bb.x + a.y * bb.y + a.z * bb.z + a.w * bb.w;
    }
    h[k] = s;
  }
  if (idx == 0) {
    float s = 0.f;
    for (int j = 0; j < Hd; ++j) s += bq[j] * bk[j];
    c[0] = s;
  }
}

// ---------- prep: A = Wq @ Wk^T, stored bf16 as As[n][k] ----------
__global__ __launch_bounds__(256) void k_prep_A(
    const float* __restrict__ Wq, const float* __restrict__ Wk,
    u16* __restrict__ Aout) {
  __shared__ u16  aslab[64 * 72];
  __shared__ u16  bslab[128 * 72];
  __shared__ float dlds[64 * 132];
  const int tid = threadIdx.x, lane = tid & 31, wave = tid >> 5;
  const int waveM = wave & 3, waveN = wave >> 2;
  const int k0 = blockIdx.x * 64, n0 = blockIdx.y * 128;

  v8f acc[4] = {vzero8(), vzero8(), vzero8(), vzero8()};
  for (int kk = 0; kk < Hd; kk += 64) {
    for (int g4 = tid; g4 < 64 * 16; g4 += 256) {          // Wq rows -> A tile
      int r = g4 >> 4, c4 = g4 & 15;
      float4 v = *(const float4*)(Wq + (size_t)(k0 + r) * Hd + kk + c4 * 4);
      st_bf4(&aslab[r * 72 + c4 * 4], v);
    }
    for (int g4 = tid; g4 < 128 * 16; g4 += 256) {         // Wk rows -> B tile [n][k]
      int r = g4 >> 4, c4 = g4 & 15;
      float4 v = *(const float4*)(Wk + (size_t)(n0 + r) * Hd + kk + c4 * 4);
      st_bf4(&bslab[r * 72 + c4 * 4], v);
    }
    __syncthreads();
    wmma_k32<72, 72>(&aslab[waveM * 16 * 72], &bslab[waveN * 64 * 72], 0,  lane, acc);
    wmma_k32<72, 72>(&aslab[waveM * 16 * 72], &bslab[waveN * 64 * 72], 32, lane, acc);
    __syncthreads();
  }
  const int g = lane >> 4, mn = lane & 15;
#pragma unroll
  for (int s = 0; s < 4; ++s)
#pragma unroll
    for (int i = 0; i < 8; ++i)
      dlds[(waveM * 16 + i + 8 * g) * 132 + waveN * 64 + s * 16 + mn] = acc[s][i];
  __syncthreads();
  for (int idx = tid; idx < 64 * 128; idx += 256) {
    int n = idx >> 6, k = idx & 63;
    Aout[(size_t)(n0 + n) * Hd + k0 + k] = f2bf(dlds[k * 132 + n]);  // transposed store
  }
}

// ---------- main: Y=Z@A chunked, logits, softmax weights, u = sum_s w_s z_s ----------
__global__ __launch_bounds__(256) void k_attn(
    const float* __restrict__ f1, const float* __restrict__ f2,
    const float* __restrict__ f3, const float* __restrict__ f4,
    const u16* __restrict__ Abf, const float* __restrict__ gv,
    const float* __restrict__ hv, const float* __restrict__ cv,
    u16* __restrict__ U) {
  __shared__ u16   ztile[64 * 1032];     // resident z M-tile, bf16 (129 KB)
  __shared__ u16   bslab[2][128 * 72];   // double-buffered A-matrix K-slabs
  __shared__ float ylds[64 * 132];       // Y chunk
  __shared__ float larr[256];
  __shared__ float zgp[64 * 4], zhp[64 * 4];
  __shared__ float zg[64], zh[64], wl[64];

  const int tid = threadIdx.x, lane = tid & 31, wave = tid >> 5;
  const int waveM = wave & 3, waveN = wave >> 2;
  const int b0 = blockIdx.x * 16;                         // 16 batches = 64 z rows
  const float* fp[4] = {f1, f2, f3, f4};

  // stage z tile (needs f32->bf16 convert, so VALU path): row r = (batch, token r%4)
  for (int g4 = tid; g4 < 64 * 256; g4 += 256) {
    int r = g4 >> 8, c4 = g4 & 255;
    const float* src = fp[r & 3] + (size_t)(b0 + (r >> 2)) * Hd + c4 * 4;
    st_bf4(&ztile[r * 1032 + c4 * 4], *(const float4*)src);
  }
  __syncthreads();

  // zg[r] = z_r . g ; zh[r] = z_r . h   (4 partials per row)
  {
    int r = tid >> 2, part = tid & 3;
    float sg = 0.f, sh = 0.f;
    for (int k = part * 256; k < part * 256 + 256; ++k) {
      float z = b2f(ztile[r * 1032 + k]);
      sg += z * gv[k];
      sh += z * hv[k];
    }
    zgp[r * 4 + part] = sg; zhp[r * 4 + part] = sh;
  }
  __syncthreads();
  if (tid < 64) {
    zg[tid] = zgp[tid*4] + zgp[tid*4+1] + zgp[tid*4+2] + zgp[tid*4+3];
    zh[tid] = zhp[tid*4] + zhp[tid*4+1] + zhp[tid*4+2] + zhp[tid*4+3];
  }
  __syncthreads();

  const int myb = tid >> 4, myt = (tid >> 2) & 3, mys = tid & 3;
  float llog = 0.f;

  for (int c = 0; c < 8; ++c) {                     // 8 N-chunks of 128 cols of A
    const int n0c = c * 128;
    const u16* src = Abf + (size_t)n0c * Hd;
    v8f acc[4] = {vzero8(), vzero8(), vzero8(), vzero8()};

    stage_slab_async(&bslab[0][0], src, tid);       // prologue: slab 0 in flight
    for (int i = 0; i < 16; ++i) {                  // 16 K-slabs of 64
      if (i + 1 < 16) {
        stage_slab_async(&bslab[(i + 1) & 1][0], src + (i + 1) * 64, tid);
        wait_async4();                              // slab i complete (in-order)
      } else {
        wait_async0();
      }
      __syncthreads();                              // publish slab i
      const int kk = i * 64;
      wmma_k32<1032, 72>(&ztile[waveM * 16 * 1032 + kk],
                         &bslab[i & 1][waveN * 64 * 72], 0,  lane, acc);
      wmma_k32<1032, 72>(&ztile[waveM * 16 * 1032 + kk],
                         &bslab[i & 1][waveN * 64 * 72], 32, lane, acc);
      __syncthreads();                              // done reading buf i -> reusable
    }
    {   // spill Y chunk to LDS
      const int g = lane >> 4, mn = lane & 15;
#pragma unroll
      for (int s = 0; s < 4; ++s)
#pragma unroll
        for (int i = 0; i < 8; ++i)
          ylds[(waveM * 16 + i + 8 * g) * 132 + waveN * 64 + s * 16 + mn] = acc[s][i];
    }
    __syncthreads();
    {   // logits[b,t,s] += y_t . z_s over this chunk (one (b,t,s) per thread)
      const float* yrow = &ylds[(myb * 4 + myt) * 132];
      const u16*  zrow = &ztile[(myb * 4 + mys) * 1032 + n0c];
      float s = 0.f;
      for (int n = 0; n < 128; ++n) s += yrow[n] * b2f(zrow[n]);
      llog += s;
    }
    __syncthreads();
  }

  llog += zg[myb * 4 + mys] + zh[myb * 4 + myt] + cv[0];
  larr[tid] = llog;
  __syncthreads();

  if (tid < 64) {   // w[b,s] = sum_t softmax_t(logits)[s]
    int b = tid >> 2, s = tid & 3;
    float w = 0.f;
    for (int t = 0; t < 4; ++t) {
      const float* lr = &larr[b * 16 + t * 4];
      float m = fmaxf(fmaxf(lr[0], lr[1]), fmaxf(lr[2], lr[3]));
      float Zt = __expf(lr[0]-m) + __expf(lr[1]-m) + __expf(lr[2]-m) + __expf(lr[3]-m);
      w += __expf(lr[s] - m) / Zt;
    }
    wl[tid] = w;
  }
  __syncthreads();

  // u[b,:] = sum_s w[b,s] * z[b,s,:]
  for (int idx = tid; idx < 16 * 1024; idx += 256) {
    int bl = idx >> 10, n = idx & 1023;
    float u = 0.f;
#pragma unroll
    for (int s = 0; s < 4; ++s)
      u += wl[bl * 4 + s] * b2f(ztile[(bl * 4 + s) * 1032 + n]);
    U[(size_t)(b0 + bl) * Hd + n] = f2bf(u);
  }
}

// ---------- main: x = u@Wv + 4bv ; out = softmax(x@Wfc + bfc) ----------
__global__ __launch_bounds__(256) void k_fc(
    const u16* __restrict__ U, const u16* __restrict__ Wvt,
    const u16* __restrict__ Wfct, const float* __restrict__ bv,
    const float* __restrict__ bfc, float* __restrict__ Xout,
    float* __restrict__ Oout) {
  __shared__ u16   utile[64 * 1032];     // resident u tile (129 KB)
  __shared__ u16   xslab[64 * 136];      // bf16 x chunk (FC A-operand)
  __shared__ u16   wvslab[2][128 * 72];  // double-buffered Wv K-slabs
  __shared__ u16   wfcslab[128 * 136];   // Wfc chunk slab
  __shared__ float outp[64 * 128];
  __shared__ float rmx[64], rsm[64];

  const int tid = threadIdx.x, lane = tid & 31, wave = tid >> 5;
  const int waveM = wave & 3, waveN = wave >> 2;
  const int r0 = blockIdx.x * 64;

  // stage u tile fully async (bf16 in ws): 32 x b128 per thread
#pragma unroll
  for (int g4 = tid; g4 < 64 * 128; g4 += 256) {
    int r = g4 >> 7, c8 = g4 & 127;
    async_ld16(&utile[r * 1032 + c8 * 8], U + (size_t)(r0 + r) * Hd + c8 * 8);
  }
  wait_async0();
  __syncthreads();

  v8f fcacc[4] = {vzero8(), vzero8(), vzero8(), vzero8()};

  for (int c = 0; c < 8; ++c) {                    // 8 N-chunks of x (= K-chunks of FC)
    const int n0c = c * 128;
    v8f acc[4] = {vzero8(), vzero8(), vzero8(), vzero8()};

    // issue the Wfc slab for this chunk first; it streams in behind the Wv slabs
#pragma unroll
    for (int g4 = tid; g4 < 128 * 16; g4 += 256) {
      int r = g4 >> 4, c8 = g4 & 15;
      async_ld16(&wfcslab[r * 136 + c8 * 8], Wfct + (size_t)r * Hd + n0c + c8 * 8);
    }

    const u16* src = Wvt + (size_t)n0c * Hd;
    stage_slab_async(&wvslab[0][0], src, tid);
    for (int i = 0; i < 16; ++i) {
      if (i + 1 < 16) {
        stage_slab_async(&wvslab[(i + 1) & 1][0], src + (i + 1) * 64, tid);
        wait_async4();
      } else {
        wait_async0();
      }
      __syncthreads();
      const int kk = i * 64;
      wmma_k32<1032, 72>(&utile[waveM * 16 * 1032 + kk],
                         &wvslab[i & 1][waveN * 64 * 72], 0,  lane, acc);
      wmma_k32<1032, 72>(&utile[waveM * 16 * 1032 + kk],
                         &wvslab[i & 1][waveN * 64 * 72], 32, lane, acc);
      __syncthreads();
    }
    {   // x = acc + 4*bv -> d_out (f32) and xslab (bf16)
      const int g = lane >> 4, mn = lane & 15;
#pragma unroll
      for (int s = 0; s < 4; ++s) {
        const int col = n0c + waveN * 64 + s * 16 + mn;
        const float bvv = 4.0f * bv[col];
#pragma unroll
        for (int i = 0; i < 8; ++i) {
          const int rl = waveM * 16 + i + 8 * g;
          const float x = acc[s][i] + bvv;
          Xout[(size_t)(r0 + rl) * Hd + col] = x;
          xslab[rl * 136 + waveN * 64 + s * 16 + mn] = f2bf(x);
        }
      }
    }
    __syncthreads();                               // xslab + wfcslab both ready
    for (int ks = 0; ks < 128; ks += 32)           // FC accumulate over this chunk
      wmma_k32<136, 136>(&xslab[waveM * 16 * 136], &wfcslab[waveN * 64 * 136], ks, lane, fcacc);
    __syncthreads();                               // before next chunk reuses slabs
  }

  {   // out_pre = fcacc + bfc -> LDS
    const int g = lane >> 4, mn = lane & 15;
#pragma unroll
    for (int s = 0; s < 4; ++s) {
      const int col = waveN * 64 + s * 16 + mn;
      const float bb = bfc[col];
#pragma unroll
      for (int i = 0; i < 8; ++i)
        outp[(waveM * 16 + i + 8 * g) * 128 + col] = fcacc[s][i] + bb;
    }
  }
  __syncthreads();
  if (tid < 64) {
    const float* row = &outp[tid * 128];
    float m = -1e30f;
    for (int n = 0; n < 128; ++n) m = fmaxf(m, row[n]);
    float s = 0.f;
    for (int n = 0; n < 128; ++n) s += __expf(row[n] - m);
    rmx[tid] = m; rsm[tid] = 1.0f / s;
  }
  __syncthreads();
  for (int idx = tid; idx < 64 * 128; idx += 256) {
    int r = idx >> 7, n = idx & 127;
    Oout[(size_t)(r0 + r) * OUTN + n] = __expf(outp[idx] - rmx[r]) * rsm[r];
  }
}

// ---------- launch ----------
extern "C" void kernel_launch(void* const* d_in, const int* in_sizes, int n_in,
                              void* d_out, int out_size, void* d_ws, size_t ws_size,
                              hipStream_t stream) {
  (void)in_sizes; (void)n_in; (void)out_size; (void)ws_size;
  const float* f1  = (const float*)d_in[0];
  const float* f2  = (const float*)d_in[1];
  const float* f3  = (const float*)d_in[2];
  const float* f4  = (const float*)d_in[3];
  const float* Wq  = (const float*)d_in[4];
  const float* bq  = (const float*)d_in[5];
  const float* Wk  = (const float*)d_in[6];
  const float* bk  = (const float*)d_in[7];
  const float* Wv  = (const float*)d_in[8];
  const float* bv  = (const float*)d_in[9];
  const float* Wfc = (const float*)d_in[10];
  const float* bfc = (const float*)d_in[11];

  char* ws = (char*)d_ws;
  const size_t OFF_A    = 0;                       // H*H bf16 = 2 MB
  const size_t OFF_WVT  = 2097152;                 // H*H bf16 = 2 MB
  const size_t OFF_WFCT = 4194304;                 // OUT*H bf16 = 256 KB
  const size_t OFF_G    = 4456448;                 // H f32
  const size_t OFF_H    = 4460544;                 // H f32
  const size_t OFF_C    = 4464640;                 // 1 f32 (padded)
  const size_t OFF_U    = 4465664;                 // B*H bf16 = 32 MB
  u16*   Abf  = (u16*)(ws + OFF_A);
  u16*   Wvt  = (u16*)(ws + OFF_WVT);
  u16*   Wfct = (u16*)(ws + OFF_WFCT);
  float* gv   = (float*)(ws + OFF_G);
  float* hv   = (float*)(ws + OFF_H);
  float* cv   = (float*)(ws + OFF_C);
  u16*   Uu   = (u16*)(ws + OFF_U);

  float* Xout = (float*)d_out;                       // [B, H]
  float* Oout = (float*)d_out + (size_t)BN * Hd;     // [B, OUT]

  k_prep_transpose<<<(Hd * Hd + Hd * OUTN) / 256, 256, 0, stream>>>(Wv, Wfc, Wvt, Wfct);
  k_prep_bias<<<(2 * Hd) / 256, 256, 0, stream>>>(Wq, Wk, bq, bk, gv, hv, cv);
  k_prep_A<<<dim3(Hd / 64, Hd / 128), 256, 0, stream>>>(Wq, Wk, Abf);
  k_attn<<<BN / 16, 256, 0, stream>>>(f1, f2, f3, f4, Abf, gv, hv, cv, Uu);
  k_fc<<<BN / 64, 256, 0, stream>>>(Uu, Wvt, Wfct, bv, bfc, Xout, Oout);
}
